// CrossAttention_4561255268507
// MI455X (gfx1250) — matrix operand order repaired and checked by reading, not compiled, not run on hardware
//
#include <hip/hip_runtime.h>

// ---------------------------------------------------------------------------
// Windowed cross-attention for MI455X (gfx1250, wave32, WMMA).
//
// K0: pack w_q / w_kv / w_proj to bf16 in WMMA A-operand lane layout (in d_ws)
// K1: per-window fused QKV projection + attention -> bf16 att (d_ws), stored in
//     global *B-operand packed* layout so K2 can feed WMMA straight from memory.
//     The attention output GEMM computes O^T = v^T x P^T so the WMMA D register
//     axis is the channel axis and lands directly in the packed layout (one
//     global_store_b128 per tile, no LDS transpose).
// K2: final channel projection (w_proj, b_proj) with 252x252 crop -> f32 d_out
//     (LDS-free: A and B operands are contiguous b128 loads)
//
// All GEMMs use v_wmma_f32_16x16x32_bf16. Operand layouts (ISA 7.12.2):
//   A (16x32 MxK): lane row = lane&15; half hb: elem e -> K = hb*16+e (contiguous, not ISA table)
//   B (32x16 KxN): lane col = lane&15; half hb: elem e -> K = hb*16+e (same order as A, not ISA table)
//   C/D f32:       lane col = lane&15; VGPR r -> M = r + 8*(lane>>4)
// Packed B tile: element (k,col) lives at [((k>>4)*16 + col)*16 + (k&15)], so a
// lane reads its 16 operand bf16 contiguously (2x b128).
//
// att global layout (bf16): image b, 16-wide spatial tile s = hn*16 + wn/16
// (4096 tiles over the padded 256x256 grid), channel c, column col = wn&15:
//   addr = (b*4096 + s)*4096 + (c>>5)*512 + (((c>>4)&1)*16 + col)*16 + (c&15)
// i.e. 8 ready-to-use 32x16 B tiles per (b,s).
//
// d_ws requirement: 128 MiB att + 512 KiB packed weights = 134,742,016 bytes.
// ---------------------------------------------------------------------------

#define C_CH   256
#define IMG    252
#define HWIMG  (252 * 252)
#define NWIN   4096
#define ATT_BYTES 134217728ull

typedef __attribute__((ext_vector_type(16))) __bf16 v16bf;
typedef __attribute__((ext_vector_type(8)))  __bf16 v8bf;
typedef __attribute__((ext_vector_type(2)))  __bf16 v2bf;
typedef __attribute__((ext_vector_type(16))) float  v16f;
typedef __attribute__((ext_vector_type(8)))  float  v8f;
typedef __attribute__((ext_vector_type(4)))  float  v4f;
typedef __attribute__((ext_vector_type(2)))  float  v2f;

__device__ __forceinline__ int a_kmap(int e, int hb) {
  return hb * 16 + e;
}

__device__ __forceinline__ v8f wmma_bf16(v16bf a, v16bf b, v8f c) {
  return __builtin_amdgcn_wmma_f32_16x16x32_bf16(false, a, false, b, (short)0, c,
                                                 false, false);
}

// Two floats -> adjacent bf16 pair (v_cvt_pk_bf16_f32).
__device__ __forceinline__ unsigned pack2(float a, float b) {
  v2f f = {a, b};
  v2bf h = __builtin_convertvector(f, v2bf);
  return __builtin_bit_cast(unsigned, h);
}

// A operand (16x32 MxK) from a row-major bf16 LDS tile (merges to 2x ds_load_b128).
__device__ __forceinline__ v16bf load_a_lds(const __bf16* base, int stride, int lane) {
  const int row = lane & 15, hb = lane >> 4;
  v16bf a;
#pragma unroll
  for (int e = 0; e < 16; ++e) a[e] = base[row * stride + a_kmap(e, hb)];
  return a;
}

// ---------------------------------------------------------------------------
// K0: weights -> bf16, A-operand packed.
// ---------------------------------------------------------------------------
__global__ void pack_w_kernel(const float* __restrict__ wq,
                              const float* __restrict__ wkv,
                              const float* __restrict__ wp,
                              unsigned short* __restrict__ dq,
                              unsigned short* __restrict__ dkv,
                              unsigned short* __restrict__ dp) {
  const int idx = blockIdx.x * 256 + threadIdx.x;   // 0 .. 262143
  const float* src;
  __bf16* dst;
  int pidx;
  if (idx < 65536)       { src = wq;  dst = (__bf16*)dq;  pidx = idx; }
  else if (idx < 196608) { src = wkv; dst = (__bf16*)dkv; pidx = idx - 65536; }
  else                   { src = wp;  dst = (__bf16*)dp;  pidx = idx - 196608; }
  const int mt = pidx >> 12, rem = pidx & 4095;
  const int kc = rem >> 9, lane = (rem >> 4) & 31, e = rem & 15;
  const int row = mt * 16 + (lane & 15);
  const int col = kc * 32 + a_kmap(e, lane >> 4);
  dst[pidx] = (__bf16)src[row * 256 + col];
}

// ---------------------------------------------------------------------------
// K1: one block per window (4096 blocks, 256 threads = 8 wave32).
// ---------------------------------------------------------------------------
__launch_bounds__(256, 1)
__global__ void win_attn_kernel(const float* __restrict__ xq,
                                const float* __restrict__ xkv,
                                const unsigned short* __restrict__ wqp_raw,
                                const unsigned short* __restrict__ wkvp_raw,
                                unsigned short* __restrict__ att_raw) {
  const __bf16* __restrict__ wqp  = reinterpret_cast<const __bf16*>(wqp_raw);
  const __bf16* __restrict__ wkvp = reinterpret_cast<const __bf16*>(wkvp_raw);
  __bf16* __restrict__ attp = reinterpret_cast<__bf16*>(att_raw);

  __shared__ __align__(32) __bf16 sXQp[4 * 512];  // x_q chunk, B-packed      4 KB
  __shared__ __align__(32) __bf16 sXKp[4 * 512];  // x_kv chunk, B-packed     4 KB
  __shared__ __align__(32) __bf16 sQ[64 * 32];    // q, A-op row-major [l][d] 4 KB
  __shared__ __align__(32) __bf16 sKp[4 * 512];   // k, B-packed (tile l>>4)  4 KB
  __shared__ __align__(32) __bf16 sVt[32 * 64];   // v^T, A-op row-major [d][m] 4 KB
  __shared__ __align__(32) float  sS[64 * 64];    // logits f32              16 KB
  __shared__ __align__(32) __bf16 sPt[8 * 512];   // P^T, B-packed (ks*4+nt)  8 KB

  const int n  = blockIdx.x;
  const int b  = n >> 10;
  const int h0 = ((n >> 5) & 31) * 8;
  const int w0 = (n & 31) * 8;

  const int t    = threadIdx.x;
  const int lane = t & 31;
  const int wv   = t >> 5;

  const float scale = 0.17677669529663689f;   // 1/sqrt(32)

  for (int head = 0; head < 8; ++head) {
    const int mo = wv & 1;    // 16-row half of the head's 32 channels
    const int nl = wv >> 1;   // 16-col tile of the 64 positions
    v8f accq = {}, acck = {}, accv = {};

    for (int kc = 0; kc < 8; ++kc) {
      // ---- stage 32ch x 64pos chunk: 2ch x 4pos patch per thread (b128 loads) ----
      {
        const int ck = (t >> 4) * 2;          // even channel-in-chunk (0..30)
        const int lq = (t & 15) * 4;          // position quad (0..60)
        const int h = h0 + (lq >> 3), w = w0 + (lq & 7);
        const bool ok = (h < IMG) && (w < IMG);  // 252%4==0 -> quad-uniform
        v4f a0 = {0.f, 0.f, 0.f, 0.f}, a1 = a0, c0 = a0, c1 = a0;
        if (ok) {
          const long off = (long)(b * C_CH + kc * 32 + ck) * HWIMG + h * IMG + w;
          a0 = *(const v4f*)(xq  + off);
          a1 = *(const v4f*)(xq  + off + HWIMG);
          c0 = *(const v4f*)(xkv + off);
          c1 = *(const v4f*)(xkv + off + HWIMG);
          if (kc < 7 && (t & 3) == 0) {       // speculative prefetch, next chunk
            __builtin_prefetch(xq  + off + 32 * HWIMG, 0, 3);
            __builtin_prefetch(xkv + off + 32 * HWIMG, 0, 3);
          }
        }
        // packed addr of (ck, lq); (ck+1)->+1, (l+1)->+16
        const int ba = (lq >> 4) * 512 + (((ck >> 4) * 16 + (lq & 15)) * 16) + (ck & 15);
        *(unsigned*)(sXQp + ba)      = pack2(a0.x, a1.x);
        *(unsigned*)(sXQp + ba + 16) = pack2(a0.y, a1.y);
        *(unsigned*)(sXQp + ba + 32) = pack2(a0.z, a1.z);
        *(unsigned*)(sXQp + ba + 48) = pack2(a0.w, a1.w);
        *(unsigned*)(sXKp + ba)      = pack2(c0.x, c1.x);
        *(unsigned*)(sXKp + ba + 16) = pack2(c0.y, c1.y);
        *(unsigned*)(sXKp + ba + 32) = pack2(c0.z, c1.z);
        *(unsigned*)(sXKp + ba + 48) = pack2(c0.w, c1.w);
      }
      __syncthreads();

      const int mtq = head * 2 + mo;
      const v16bf aq = *(const v16bf*)(wqp  + ((mtq * 8 + kc) * 32 + lane) * 16);
      const v16bf ak = *(const v16bf*)(wkvp + ((mtq * 8 + kc) * 32 + lane) * 16);
      const v16bf av = *(const v16bf*)(wkvp + (((16 + mtq) * 8 + kc) * 32 + lane) * 16);
      const v16bf bq = *(const v16bf*)(sXQp + nl * 512 + lane * 16);
      const v16bf bk = *(const v16bf*)(sXKp + nl * 512 + lane * 16);

      accq = wmma_bf16(aq, bq, accq);
      acck = wmma_bf16(ak, bk, acck);
      accv = wmma_bf16(av, bk, accv);
      __syncthreads();
    }

    // ---- scatter q/k/v tiles into attention operand layouts ----
    {
      const int nn = lane & 15, hb = lane >> 4;
      const int l = nl * 16 + nn;
      // d = mo*16 + hb*8 + r : r-contiguous -> single b128 store for q/k
      const int qb = l * 32 + mo * 16 + hb * 8;                // sQ row-major [l][d]
      const int kb = nl * 512 + (mo * 16 + nn) * 16 + hb * 8;  // sKp packed
      *reinterpret_cast<v8bf*>(sQ  + qb) = __builtin_convertvector(accq, v8bf);
      *reinterpret_cast<v8bf*>(sKp + kb) = __builtin_convertvector(acck, v8bf);
      // sVt row-major [d][m=l]: v tile is already (d x l) oriented
      const int vb = (mo * 16 + hb * 8) * 64 + l;
#pragma unroll
      for (int r = 0; r < 8; ++r) sVt[vb + r * 64] = (__bf16)accv[r];
    }
    __syncthreads();

    // ---- S = (q k) * scale : 16 tiles, K = hd = 32 in one WMMA each ----
#pragma unroll
    for (int it = 0; it < 2; ++it) {
      const int ti = wv + it * 8;
      const int tr = ti >> 2, tc = ti & 3;
      const v16bf a  = load_a_lds(sQ + tr * 16 * 32, 32, lane);
      const v16bf bb = *(const v16bf*)(sKp + tc * 512 + lane * 16);
      v8f s = {};
      s = wmma_bf16(a, bb, s);
      const int nn = lane & 15, hb = lane >> 4;
#pragma unroll
      for (int r = 0; r < 8; ++r)
        sS[(tr * 16 + r + 8 * hb) * 64 + tc * 16 + nn] = s[r] * scale;
    }
    __syncthreads();

    // ---- row softmax; write P^T directly in packed-B layout ----
    // element (m, l): tile (m>>5)*4 + (l>>4), addr ((m&31)>>4)*16+(l&15))*16 + (m&15)
    if (t < 64) {
      float m = -1e30f;
      for (int j = 0; j < 64; ++j) m = fmaxf(m, sS[t * 64 + j]);
      float sum = 0.0f;
      for (int j = 0; j < 64; ++j) sum += __expf(sS[t * 64 + j] - m);
      const float inv = 1.0f / sum;
      const int colr = t & 15, nt = t >> 4;
#pragma unroll
      for (int g = 0; g < 4; ++g) {       // m = g*16 + jj : one contiguous 32B run
        v16f pv;
#pragma unroll
        for (int jj = 0; jj < 16; ++jj)
          pv[jj] = __expf(sS[t * 64 + g * 16 + jj] - m) * inv;
        *reinterpret_cast<v16bf*>(sPt + ((g >> 1) * 4 + nt) * 512 +
                                  ((g & 1) * 16 + colr) * 16) =
            __builtin_convertvector(pv, v16bf);
      }
    }
    __syncthreads();

    // ---- O^T = v^T P^T : D register axis = channel -> direct packed store ----
    {
      const int mt = wv & 1, nt = wv >> 1;   // d-tile, l-tile
      v8f o = {};
#pragma unroll
      for (int ks = 0; ks < 2; ++ks) {
        const v16bf a  = load_a_lds(sVt + mt * 16 * 64 + ks * 32, 64, lane);
        const v16bf bb = *(const v16bf*)(sPt + (ks * 4 + nt) * 512 + lane * 16);
        o = wmma_bf16(a, bb, o);
      }
      const int nn = lane & 15, hb = lane >> 4;
      const int l  = nt * 16 + nn;
      const int hn = h0 + (l >> 3);
      const int col = (w0 & 8) + (l & 7);
      // c = head*32 + mt*16 + 8*hb + r ; e = c&15 = 8*hb + r (r-contiguous)
      const long gaddr = ((long)(b * 4096 + hn * 16 + (w0 >> 4))) * 4096 +
                         head * 512 + (mt * 16 + col) * 16 + 8 * hb;
      *reinterpret_cast<v8bf*>(attp + gaddr) = __builtin_convertvector(o, v8bf);
    }
    __syncthreads();   // LDS reused by next head
  }
}

// ---------------------------------------------------------------------------
// K2: final projection. LDS-free: att is already 8 packed B tiles per (b,s).
// ---------------------------------------------------------------------------
__launch_bounds__(256, 1)
__global__ void proj_kernel(const unsigned short* __restrict__ att_raw,
                            const unsigned short* __restrict__ wpp_raw,
                            const float* __restrict__ bp,
                            float* __restrict__ out) {
  const __bf16* __restrict__ attp = reinterpret_cast<const __bf16*>(att_raw);
  const __bf16* __restrict__ wpp  = reinterpret_cast<const __bf16*>(wpp_raw);

  const int tileId = blockIdx.x;          // 0 .. 4*4096-1
  const int b = tileId >> 12;
  const int s = tileId & 4095;
  const int hn  = s >> 4;
  const int wn0 = (s & 15) * 16;
  if (hn >= IMG) return;                  // fully-padded rows

  const int t = threadIdx.x, lane = t & 31, wv = t >> 5;

  const long bbase = ((long)(b * 4096 + s)) * 4096 + lane * 16;
  v8f acc0 = {}, acc1 = {};
#pragma unroll
  for (int kc = 0; kc < 8; ++kc) {
    const v16bf bb = *(const v16bf*)(attp + bbase + kc * 512);
    const v16bf a0 = *(const v16bf*)(wpp + ((wv * 8 + kc) * 32 + lane) * 16);
    const v16bf a1 = *(const v16bf*)(wpp + (((wv + 8) * 8 + kc) * 32 + lane) * 16);
    acc0 = wmma_bf16(a0, bb, acc0);
    acc1 = wmma_bf16(a1, bb, acc1);
  }

  const int nn = lane & 15, hb = lane >> 4;
  const int wn = wn0 + nn;
  const v4f b0a = *(const v4f*)(bp + wv * 16 + 8 * hb);
  const v4f b0b = *(const v4f*)(bp + wv * 16 + 8 * hb + 4);
  const v4f b1a = *(const v4f*)(bp + (wv + 8) * 16 + 8 * hb);
  const v4f b1b = *(const v4f*)(bp + (wv + 8) * 16 + 8 * hb + 4);
  if (wn < IMG) {
    const long obase = (long)b * C_CH * HWIMG + (long)hn * IMG + wn;
#pragma unroll
    for (int r = 0; r < 8; ++r) {
      const int o0 = wv * 16 + 8 * hb + r;
      const int o1 = (wv + 8) * 16 + 8 * hb + r;
      out[obase + (long)o0 * HWIMG] = acc0[r] + ((r < 4) ? b0a[r] : b0b[r - 4]);
      out[obase + (long)o1 * HWIMG] = acc1[r] + ((r < 4) ? b1a[r] : b1b[r - 4]);
    }
  }
}

// ---------------------------------------------------------------------------
extern "C" void kernel_launch(void* const* d_in, const int* in_sizes, int n_in,
                              void* d_out, int out_size, void* d_ws, size_t ws_size,
                              hipStream_t stream) {
  const float* xq  = (const float*)d_in[0];
  const float* xkv = (const float*)d_in[1];
  const float* wq  = (const float*)d_in[2];
  const float* wkv = (const float*)d_in[3];
  const float* wp  = (const float*)d_in[4];
  const float* bp  = (const float*)d_in[5];
  float* out = (float*)d_out;

  // d_ws layout: [att bf16 128MiB][wq_pack 128KiB][wkv_pack 256KiB][wp_pack 128KiB]
  unsigned short* att  = (unsigned short*)d_ws;
  unsigned short* wqp  = (unsigned short*)((char*)d_ws + ATT_BYTES);
  unsigned short* wkvp = (unsigned short*)((char*)d_ws + ATT_BYTES + 131072);
  unsigned short* wpp  = (unsigned short*)((char*)d_ws + ATT_BYTES + 393216);

  hipLaunchKernelGGL(pack_w_kernel, dim3(1024), dim3(256), 0, stream,
                     wq, wkv, wp, wqp, wkvp, wpp);
  hipLaunchKernelGGL(win_attn_kernel, dim3(NWIN), dim3(256), 0, stream,
                     xq, xkv, wqp, wkvp, att);
  hipLaunchKernelGGL(proj_kernel, dim3(4 * 4096), dim3(256), 0, stream,
                     att, wpp, bp, out);
}
